// cudaH_18339510354596
// MI455X (gfx1250) — compile-verified, matrix operand order checked by reading, not run on hardware
//
#include <hip/hip_runtime.h>
#include <hip/hip_bf16.h>
#include <math.h>

// ---------------------------------------------------------------------------
// MI455X (gfx1250) fused cosine-similarity / max / entropy pipeline.
//   C = normalize_rows(ex) @ normalize_rows(ey)^T   (8192 x 8192, K=768)
//   H1 = entropy(rowmax(C)), H2 = entropy(colmax(C))
// Compute-bound (103 GFLOP vs 50MB traffic) -> bf16 WMMA with hi/lo split
// (3x v_wmma_f32_16x16x32_bf16 per logical f32 MAC tile) for ~fp32 accuracy.
// C never materialized: max reductions fused into the GEMM epilogue.
// Double-buffered LDS + CDNA5 async global->LDS copies
// (global_load_async_to_lds_b128 / s_wait_asynccnt) to overlap HBM latency
// with the WMMA stream; one barrier per k-stage.
// ---------------------------------------------------------------------------

typedef __bf16 bf16_t;
typedef bf16_t bf16x16 __attribute__((ext_vector_type(16)));
typedef float  f32x8   __attribute__((ext_vector_type(8)));
typedef int    v4i     __attribute__((vector_size(4 * sizeof(int))));

#define NX   8192
#define NY   8192
#define DIM  768
#define BM   128
#define BN   128
#define BK   32
#define ROW_U4 5            // uint4 per LDS row: 4 data (32 bf16) + 1 pad
#define BUFSZ (4 * BM * ROW_U4)   // uint4 per stage buffer (40 KB)

#if defined(__AMDGCN__) && __has_builtin(__builtin_amdgcn_global_load_async_to_lds_b128) && \
    __has_builtin(__builtin_amdgcn_s_wait_asynccnt)
#define ASYNC_COPY 1
#else
#define ASYNC_COPY 0
#endif

#if ASYNC_COPY
#define CP16(dst, src)                                                        \
  __builtin_amdgcn_global_load_async_to_lds_b128(                             \
      (__attribute__((address_space(1))) v4i*)(src),                          \
      (__attribute__((address_space(3))) v4i*)(dst), 0, 0)
#define WAIT_COPY() __builtin_amdgcn_s_wait_asynccnt(0)
#else
#define CP16(dst, src) (*(dst) = *(src))
#define WAIT_COPY()
#endif

// Order-preserving float -> uint key so atomicMax(u32) implements float max
// (cosine values can be negative).
__device__ __forceinline__ unsigned fkey(float f) {
  unsigned u = __float_as_uint(f);
  return (u & 0x80000000u) ? ~u : (u | 0x80000000u);
}
__device__ __forceinline__ float funkey(unsigned k) {
  unsigned u = (k & 0x80000000u) ? (k & 0x7FFFFFFFu) : ~k;
  return __uint_as_float(u);
}

__global__ void init_max_kernel(unsigned* __restrict__ keys, int n) {
  int i = blockIdx.x * blockDim.x + threadIdx.x;
  if (i < n) keys[i] = 0u;   // decodes below any real cosine key
}

// One block per row (ex rows first, then ey rows). Normalizes in f32, then
// splits each value into bf16 hi + bf16 lo (lo = x - (float)hi).
__global__ __launch_bounds__(256) void normalize_split_kernel(
    const float* __restrict__ ex, const float* __restrict__ ey,
    bf16_t* __restrict__ exh, bf16_t* __restrict__ exl,
    bf16_t* __restrict__ eyh, bf16_t* __restrict__ eyl) {
  int row = blockIdx.x;
  const float* src;
  bf16_t *dh, *dl;
  if (row < NX) {
    src = ex + (size_t)row * DIM;
    dh = exh + (size_t)row * DIM;
    dl = exl + (size_t)row * DIM;
  } else {
    int r = row - NX;
    src = ey + (size_t)r * DIM;
    dh = eyh + (size_t)r * DIM;
    dl = eyl + (size_t)r * DIM;
  }
  int t = threadIdx.x;
  float x0 = src[t], x1 = src[t + 256], x2 = src[t + 512];
  __shared__ float red[256];
  red[t] = x0 * x0 + x1 * x1 + x2 * x2;
  __syncthreads();
  for (int off = 128; off > 0; off >>= 1) {
    if (t < off) red[t] += red[t + off];
    __syncthreads();
  }
  float inv = 1.0f / fmaxf(sqrtf(red[0]), 1e-8f);
  float y;
  y = x0 * inv; { bf16_t h = (bf16_t)y; dh[t]       = h; dl[t]       = (bf16_t)(y - (float)h); }
  y = x1 * inv; { bf16_t h = (bf16_t)y; dh[t + 256] = h; dl[t + 256] = (bf16_t)(y - (float)h); }
  y = x2 * inv; { bf16_t h = (bf16_t)y; dh[t + 512] = h; dl[t + 512] = (bf16_t)(y - (float)h); }
}

// 128x128 C tile per block, 8 waves; each wave owns 64x32 (4x2 WMMA tiles).
// Per k-step each logical tile does 3 bf16 WMMAs: Ah*Bh + Ah*Bl + Al*Bh.
__global__ __launch_bounds__(256) void gemm_max_kernel(
    const bf16_t* __restrict__ exh, const bf16_t* __restrict__ exl,
    const bf16_t* __restrict__ eyh, const bf16_t* __restrict__ eyl,
    unsigned* __restrict__ rowmax, unsigned* __restrict__ colmax) {
  __shared__ uint4 lds[2 * BUFSZ];   // 80 KB double buffer: [Ah|Al|Bh|Bl] x2

  const int tid   = threadIdx.x;
  const int lane  = tid & 31;
  const int w     = tid >> 5;   // 0..7
  const int wr    = w >> 2;     // 0..1  -> 64-row group
  const int wc    = w & 3;      // 0..3  -> 32-col group
  const int mrow  = lane & 15;  // tile row (A) / tile col (B) per ISA layout
  const int khalf = lane >> 4;  // selects K halves per ISA layout
  const int mbase = blockIdx.y * BM;
  const int nbase = blockIdx.x * BN;

  const uint4* gAh = (const uint4*)exh + (size_t)mbase * (DIM / 8);
  const uint4* gAl = (const uint4*)exl + (size_t)mbase * (DIM / 8);
  const uint4* gBh = (const uint4*)eyh + (size_t)nbase * (DIM / 8);
  const uint4* gBl = (const uint4*)eyl + (size_t)nbase * (DIM / 8);

  // Issue the 16B chunks of one k-stage into LDS buffer `bufsel`.
  auto copyStage = [&](int bufsel, int s) {
    uint4* b = lds + bufsel * BUFSZ;
    const int kq = s * (BK / 8);   // uint4 offset within a matrix row
#pragma unroll
    for (int p = 0; p < 2; ++p) {
      int slot = tid + p * 256;    // 512 chunk slots per matrix
      int r = slot >> 2;
      int c = slot & 3;
      int g = r * (DIM / 8) + kq + c;
      int l = r * ROW_U4 + c;
      CP16(&b[l],                 &gAh[g]);
      CP16(&b[BUFSZ / 4 + l],     &gAl[g]);
      CP16(&b[BUFSZ / 2 + l],     &gBh[g]);
      CP16(&b[3 * BUFSZ / 4 + l], &gBl[g]);
    }
  };

  const f32x8 zero = {0.f, 0.f, 0.f, 0.f, 0.f, 0.f, 0.f, 0.f};
  f32x8 acc[4][2];
#pragma unroll
  for (int i = 0; i < 4; ++i)
#pragma unroll
    for (int j = 0; j < 2; ++j) acc[i][j] = zero;

  const int NSTAGE = DIM / BK;
  copyStage(0, 0);               // prime the pipeline

  for (int s = 0; s < NSTAGE; ++s) {
    const int cur = s & 1;
    WAIT_COPY();                 // own async copies for stage s have landed
    __syncthreads();             // everyone's copies for stage s have landed
    if (s + 1 < NSTAGE) copyStage(cur ^ 1, s + 1);  // overlap with WMMAs below

    const uint4* sAh = lds + cur * BUFSZ;
    const uint4* sAl = sAh + BUFSZ / 4;
    const uint4* sBh = sAh + BUFSZ / 2;
    const uint4* sBl = sAh + 3 * BUFSZ / 4;

    // B fragments (hi and lo) for this wave's two 16-col tiles
    bf16x16 Bh[2], Bl[2];
#pragma unroll
    for (int j = 0; j < 2; ++j) {
      int col = wc * 32 + j * 16 + mrow;
      union { bf16x16 v; uint4 q[2]; } u;
      u.q[0] = sBh[col * ROW_U4 + khalf];
      u.q[1] = sBh[col * ROW_U4 + 2 + khalf];
      Bh[j] = u.v;
      u.q[0] = sBl[col * ROW_U4 + khalf];
      u.q[1] = sBl[col * ROW_U4 + 2 + khalf];
      Bl[j] = u.v;
    }
#pragma unroll
    for (int i = 0; i < 4; ++i) {
      int rowi = wr * 64 + i * 16 + mrow;
      union { bf16x16 v; uint4 q[2]; } ua, ul;
      ua.q[0] = sAh[rowi * ROW_U4 + khalf];
      ua.q[1] = sAh[rowi * ROW_U4 + 2 + khalf];
      ul.q[0] = sAl[rowi * ROW_U4 + khalf];
      ul.q[1] = sAl[rowi * ROW_U4 + 2 + khalf];
#pragma unroll
      for (int j = 0; j < 2; ++j) {
        acc[i][j] = __builtin_amdgcn_wmma_f32_16x16x32_bf16(
            false, ua.v, false, Bh[j], (short)0, acc[i][j], false, false);
        acc[i][j] = __builtin_amdgcn_wmma_f32_16x16x32_bf16(
            false, ua.v, false, Bl[j], (short)0, acc[i][j], false, false);
        acc[i][j] = __builtin_amdgcn_wmma_f32_16x16x32_bf16(
            false, ul.v, false, Bh[j], (short)0, acc[i][j], false, false);
      }
    }
  }

  // ---- fused epilogue: row-max (over N) and col-max (over M) ----
  // C/D layout: VGPR r holds row (i*16 + khalf*8 + r), col (lane&15).
#pragma unroll
  for (int i = 0; i < 4; ++i) {
#pragma unroll
    for (int r = 0; r < 8; ++r) {
      float t = fmaxf(acc[i][0][r], acc[i][1][r]);
      t = fmaxf(t, __shfl_xor(t, 1, 32));
      t = fmaxf(t, __shfl_xor(t, 2, 32));
      t = fmaxf(t, __shfl_xor(t, 4, 32));
      t = fmaxf(t, __shfl_xor(t, 8, 32));
      if (mrow == 0) {
        int grow = mbase + wr * 64 + i * 16 + khalf * 8 + r;
        atomicMax(&rowmax[grow], fkey(t));
      }
    }
  }
#pragma unroll
  for (int j = 0; j < 2; ++j) {
    float cm = -3.0f;
#pragma unroll
    for (int i = 0; i < 4; ++i) {
      float t = acc[i][j][0];
#pragma unroll
      for (int r = 1; r < 8; ++r) t = fmaxf(t, acc[i][j][r]);
      t = fmaxf(t, __shfl_xor(t, 16, 32));  // rows 0-7 vs 8-15 of the tile
      cm = fmaxf(cm, t);
    }
    if (lane < 16) atomicMax(&colmax[nbase + wc * 32 + j * 16 + lane], fkey(cm));
  }
}

// Single block: HalfNormal log-prob + entropy sums over the 8192-long
// row/col max vectors -> two scalars.
__global__ __launch_bounds__(256) void entropy_kernel(
    const unsigned* __restrict__ rowmax, const unsigned* __restrict__ colmax,
    float* __restrict__ out) {
  const float c0     = -0.0026478013305178f;  // 0.5*log(2/pi) - log(0.8)
  const float inv2s2 = 0.78125f;              // 1/(2*sigma^2), sigma=0.8
  const float invs   = 1.25f;                 // 1/sigma
  int t = threadIdx.x;
  float s1 = 0.f, s2 = 0.f;
  for (int i = t; i < NX; i += 256) {
    float x = 1.0f + funkey(rowmax[i]);
    float L = c0 - x * x * inv2s2;
    s1 -= expf(L) * invs * L;
  }
  for (int i = t; i < NY; i += 256) {
    float x = 1.0f + funkey(colmax[i]);
    float L = c0 - x * x * inv2s2;
    s2 -= expf(L) * invs * L;
  }
  __shared__ float r1[256], r2[256];
  r1[t] = s1;
  r2[t] = s2;
  __syncthreads();
  for (int off = 128; off > 0; off >>= 1) {
    if (t < off) { r1[t] += r1[t + off]; r2[t] += r2[t + off]; }
    __syncthreads();
  }
  if (t == 0) { out[0] = r1[0]; out[1] = r2[0]; }
}

extern "C" void kernel_launch(void* const* d_in, const int* in_sizes, int n_in,
                              void* d_out, int out_size, void* d_ws, size_t ws_size,
                              hipStream_t stream) {
  (void)in_sizes; (void)n_in; (void)out_size; (void)ws_size;
  const float* ex = (const float*)d_in[0];
  const float* ey = (const float*)d_in[1];
  char* ws = (char*)d_ws;

  const size_t mat_bytes = (size_t)NX * DIM * sizeof(bf16_t);  // 12.58 MB each
  bf16_t* exh = (bf16_t*)(ws);
  bf16_t* exl = (bf16_t*)(ws + mat_bytes);
  bf16_t* eyh = (bf16_t*)(ws + 2 * mat_bytes);
  bf16_t* eyl = (bf16_t*)(ws + 3 * mat_bytes);
  unsigned* rowmax = (unsigned*)(ws + 4 * mat_bytes);
  unsigned* colmax = rowmax + NX;
  float* out = (float*)d_out;

  init_max_kernel<<<(NX + NY + 255) / 256, 256, 0, stream>>>(rowmax, NX + NY);
  normalize_split_kernel<<<NX + NY, 256, 0, stream>>>(ex, ey, exh, exl, eyh, eyl);
  gemm_max_kernel<<<dim3(NY / BN, NX / BM), 256, 0, stream>>>(exh, exl, eyh, eyl,
                                                              rowmax, colmax);
  entropy_kernel<<<1, 256, 0, stream>>>(rowmax, colmax, out);
}
